// MUAttentionLayer_944892805621
// MI455X (gfx1250) — matrix operand order repaired and checked by reading, not compiled, not run on hardware
//
#include <hip/hip_runtime.h>

typedef __attribute__((ext_vector_type(16))) _Float16 v16h;
typedef __attribute__((ext_vector_type(8)))  float    v8f;

#define T_SEQ 2048
#define DM    128
#define NH    4
#define DH    32
#define NTOK  8192          // B * T
#define KSTRIDE 40          // f16 units, 80B rows: 16B aligned, conflict-free banks
#define VSTRIDE 72          // f16 units, 144B rows
#define PSTRIDE 72

union Frag {
  v16h     h;
  uint4    u[2];
  _Float16 e[16];
};

// ---- xor-butterfly lane reductions within 16-lane rows (VALU, no LDS) ---------
// v_permlane16_b32 select patterns for lane^1, ^3, ^4, ^8
#define XS0_1 0x67452301u
#define XS1_1 0xEFCDAB89u
#define XS0_2 0x45670123u
#define XS1_2 0xCDEF89ABu
#define XS0_4 0x32107654u
#define XS1_4 0xBA98FEDCu
#define XS0_8 0xFEDCBA98u
#define XS1_8 0x76543210u

#if __has_builtin(__builtin_amdgcn_permlane16)
__device__ __forceinline__ float plane16(float x, unsigned s0, unsigned s1) {
  const unsigned u = __float_as_uint(x);
  return __uint_as_float(__builtin_amdgcn_permlane16(u, u, s0, s1, false, false));
}
__device__ __forceinline__ float max16(float x) {
  x = fmaxf(x, plane16(x, XS0_1, XS1_1));
  x = fmaxf(x, plane16(x, XS0_2, XS1_2));
  x = fmaxf(x, plane16(x, XS0_4, XS1_4));
  x = fmaxf(x, plane16(x, XS0_8, XS1_8));
  return x;
}
__device__ __forceinline__ float sum16(float x) {
  x += plane16(x, XS0_1, XS1_1);
  x += plane16(x, XS0_2, XS1_2);
  x += plane16(x, XS0_4, XS1_4);
  x += plane16(x, XS0_8, XS1_8);
  return x;
}
#else
__device__ __forceinline__ float max16(float x) {
  x = fmaxf(x, __shfl_xor(x, 1, 32));
  x = fmaxf(x, __shfl_xor(x, 2, 32));
  x = fmaxf(x, __shfl_xor(x, 4, 32));
  x = fmaxf(x, __shfl_xor(x, 8, 32));
  return x;
}
__device__ __forceinline__ float sum16(float x) {
  x += __shfl_xor(x, 1, 32);
  x += __shfl_xor(x, 2, 32);
  x += __shfl_xor(x, 4, 32);
  x += __shfl_xor(x, 8, 32);
  return x;
}
#endif

// ---- async global->LDS (ASYNCcnt path), with register-path fallback -----------
#if __has_builtin(__builtin_amdgcn_global_load_async_to_lds_b128)
#define ASYNC_LDS 1
typedef int v4i_g __attribute__((vector_size(4 * sizeof(int))));
typedef __attribute__((address_space(1))) v4i_g gv4i;
typedef __attribute__((address_space(3))) v4i_g lv4i;
__device__ __forceinline__ void async_copy_b128(const void* g, void* l) {
  __builtin_amdgcn_global_load_async_to_lds_b128(
      (gv4i*)(unsigned long long)g, (lv4i*)l, 0, 0);
}
__device__ __forceinline__ void wait_async0() {
#if __has_builtin(__builtin_amdgcn_s_wait_asynccnt)
  __builtin_amdgcn_s_wait_asynccnt(0);
#else
  asm volatile("s_wait_asynccnt 0" ::: "memory");
#endif
}
#endif

// ---------------- Kernel 1: QKV projection (x[16] @ W[16x128]) ----------------
__global__ __launch_bounds__(128) void qkv_proj(
    const float* __restrict__ M,
    const float* __restrict__ Wq, const float* __restrict__ bq,
    const float* __restrict__ Wk, const float* __restrict__ bk,
    const float* __restrict__ Wv, const float* __restrict__ bv,
    _Float16* __restrict__ Qb, _Float16* __restrict__ Kb, _Float16* __restrict__ Vb)
{
  __shared__ float xs[16];
  const int tk  = blockIdx.x;
  const int tid = threadIdx.x;
  if (tid < 16) xs[tid] = M[tk * 16 + tid];
  __syncthreads();
  const int d = tid;
  float q = bq[d], k = bk[d], v = bv[d];
#pragma unroll
  for (int i = 0; i < 16; ++i) {
    const float x = xs[i];
    q = fmaf(x, Wq[i * DM + d], q);
    k = fmaf(x, Wk[i * DM + d], k);
    v = fmaf(x, Wv[i * DM + d], v);
  }
  const int b = tk >> 11, t = tk & 2047;
  const int h = d >> 5,  dd = d & 31;
  const size_t idx = ((size_t)((b * NH + h) * T_SEQ + t)) * DH + dd;
  Qb[idx] = (_Float16)(q * 0.17677669529663687f);   // fold 1/sqrt(32) into Q
  Kb[idx] = (_Float16)k;
  Vb[idx] = (_Float16)v;
}

// ---------------- Kernel 2: flash attention, v_wmma_f32_16x16x32_f16 -----------
// grid = 16 heads * 32 q-chunks = 512 blocks; 128 threads = 4 waves;
// each wave owns a 16-query tile, block streams 64-key chunks via LDS.
__global__ __launch_bounds__(128) void flash_attn(
    const _Float16* __restrict__ Qb, const _Float16* __restrict__ Kb,
    const _Float16* __restrict__ Vb, float* __restrict__ ctx)
{
  __shared__ __align__(16) _Float16 Ksh[64 * KSTRIDE];       // K chunk [key][d]
  __shared__ __align__(16) _Float16 Vsh[32 * VSTRIDE];       // V chunk transposed [d][key]
  __shared__ __align__(16) _Float16 Psh[4 * 16 * PSTRIDE];   // per-wave probs [q][key]

  const int bh   = blockIdx.x >> 5;
  const int qc   = blockIdx.x & 31;
  const int b    = bh >> 2, h = bh & 3;
  const int wave = threadIdx.x >> 5;
  const int lane = threadIdx.x & 31;
  const int lo   = lane & 15;
  const int kh   = lane >> 4;
  const int q0   = qc * 64 + wave * 16;

  const _Float16* __restrict__ Qh = Qb + (size_t)bh * T_SEQ * DH;
  const _Float16* __restrict__ Kh = Kb + (size_t)bh * T_SEQ * DH;
  const _Float16* __restrict__ Vh = Vb + (size_t)bh * T_SEQ * DH;

  // A-fragment of Q: lane holds row (q0+lo), k = kh*16+[0..15]
  Frag qa;
  {
    const _Float16* qr = Qh + (size_t)(q0 + lo) * DH;
    qa.u[0] = *(const uint4*)(qr + kh * 16);
    qa.u[1] = *(const uint4*)(qr + kh * 16 + 8);
  }

  // all-ones B fragment: P x ones = per-row sums, replicated across columns
  Frag ones;
#pragma unroll
  for (int i = 0; i < 16; ++i) ones.e[i] = (_Float16)1.0f;

  float O0[8], O1[8], Mr[8], Lr[8];
#pragma unroll
  for (int v = 0; v < 8; ++v) { O0[v] = 0.f; O1[v] = 0.f; Mr[v] = -3.0e38f; Lr[v] = 0.f; }

  _Float16* __restrict__ Pw = Psh + wave * 16 * PSTRIDE;
  const int sr = threadIdx.x >> 1;   // staging row 0..63
  const int sh = threadIdx.x & 1;    // staging half of d

  for (int kc = 0; kc < 32; ++kc) {
    __syncthreads();                 // previous iteration's LDS reads done
    {
      // stage K row-major (async direct-to-LDS if available)
      const _Float16* kr = Kh + (size_t)(kc * 64 + sr) * DH + sh * 16;
      _Float16* kd = Ksh + sr * KSTRIDE + sh * 16;
#ifdef ASYNC_LDS
      async_copy_b128(kr, kd);
      async_copy_b128(kr + 8, kd + 8);
#else
      uint4 k0 = *(const uint4*)kr;
      uint4 k1 = *(const uint4*)(kr + 8);
      *(uint4*)kd       = k0;
      *(uint4*)(kd + 8) = k1;
#endif
      // stage V transposed [d][key] (register path: transpose on the fly)
      const _Float16* vr = Vh + (size_t)(kc * 64 + sr) * DH + sh * 16;
      Frag vf;
      vf.u[0] = *(const uint4*)vr;
      vf.u[1] = *(const uint4*)(vr + 8);
#pragma unroll
      for (int e = 0; e < 16; ++e)
        Vsh[(sh * 16 + e) * VSTRIDE + sr] = vf.e[e];
      if (kc + 1 < 32) {
        __builtin_prefetch(Kh + (size_t)((kc + 1) * 64 + sr) * DH, 0, 3);
        __builtin_prefetch(Vh + (size_t)((kc + 1) * 64 + sr) * DH, 0, 3);
      }
#ifdef ASYNC_LDS
      wait_async0();
#endif
    }
    __syncthreads();

    // S = Q * K^T : 4 sub-tiles of 16 keys
    float S[4][8];
#pragma unroll
    for (int sb = 0; sb < 4; ++sb) {
      Frag kb;  // B-fragment: lane = key sb*16+lo, k = kh*16 + [0..15]
      const _Float16* kr = Ksh + (sb * 16 + lo) * KSTRIDE + kh * 16;
      kb.u[0] = *(const uint4*)kr;
      kb.u[1] = *(const uint4*)(kr + 8);
      v8f c = {};
      c = __builtin_amdgcn_wmma_f32_16x16x32_f16(false, qa.h, false, kb.h,
                                                 (short)0, c, false, false);
#pragma unroll
      for (int v = 0; v < 8; ++v) S[sb][v] = c[v];
    }

    // online softmax; C layout: row = kh*8+v, col = lo
#pragma unroll
    for (int v = 0; v < 8; ++v) {
      float nm = fmaxf(fmaxf(S[0][v], S[1][v]), fmaxf(S[2][v], S[3][v]));
      nm = max16(nm);
      const float Mn    = fmaxf(Mr[v], nm);
      const float alpha = __expf(Mr[v] - Mn);
      Mr[v] = Mn; Lr[v] *= alpha; O0[v] *= alpha; O1[v] *= alpha;
      _Float16* prow = Pw + (kh * 8 + v) * PSTRIDE;
#pragma unroll
      for (int sb = 0; sb < 4; ++sb) {
        const float p = __expf(S[sb][v] - Mn);
        prow[sb * 16 + lo] = (_Float16)p;   // C->A transpose via LDS
      }
    }

    // O += P*V and L += P*ones : two 32-key halves, two 16-col d_head blocks
    v8f cl;
#pragma unroll
    for (int v = 0; v < 8; ++v) cl[v] = Lr[v];

#pragma unroll
    for (int hc = 0; hc < 2; ++hc) {
      Frag pa;  // A-fragment of P: lane row lo, k = hc*32 + kh*16 + [0..15]
      const _Float16* pr = Pw + lo * PSTRIDE + hc * 32;
      pa.u[0] = *(const uint4*)(pr + kh * 16);
      pa.u[1] = *(const uint4*)(pr + kh * 16 + 8);

      // row sums via WMMA (replaces lane-shuffle reduction)
      cl = __builtin_amdgcn_wmma_f32_16x16x32_f16(false, pa.h, false, ones.h,
                                                  (short)0, cl, false, false);

      Frag vb;  // B-fragment: lane col d = lo (+16), k = hc*32 + kh*16 + [0..15]
      const _Float16* vr0 = Vsh + lo * VSTRIDE + hc * 32 + kh * 16;
      vb.u[0] = *(const uint4*)vr0;
      vb.u[1] = *(const uint4*)(vr0 + 8);
      v8f c;
#pragma unroll
      for (int v = 0; v < 8; ++v) c[v] = O0[v];
      c = __builtin_amdgcn_wmma_f32_16x16x32_f16(false, pa.h, false, vb.h,
                                                 (short)0, c, false, false);
#pragma unroll
      for (int v = 0; v < 8; ++v) O0[v] = c[v];

      const _Float16* vr1 = Vsh + (16 + lo) * VSTRIDE + hc * 32 + kh * 16;
      vb.u[0] = *(const uint4*)vr1;
      vb.u[1] = *(const uint4*)(vr1 + 8);
#pragma unroll
      for (int v = 0; v < 8; ++v) c[v] = O1[v];
      c = __builtin_amdgcn_wmma_f32_16x16x32_f16(false, pa.h, false, vb.h,
                                                 (short)0, c, false, false);
#pragma unroll
      for (int v = 0; v < 8; ++v) O1[v] = c[v];
    }
#pragma unroll
    for (int v = 0; v < 8; ++v) Lr[v] = cl[v];
  }

  // epilogue: O / L, write ctx[b][t][h*32+d]
#pragma unroll
  for (int v = 0; v < 8; ++v) {
    const float inv = 1.0f / Lr[v];
    const int   q   = q0 + kh * 8 + v;
    float* dst = ctx + ((size_t)(b * T_SEQ + q)) * DM + h * DH;
    dst[lo]      = O0[v] * inv;
    dst[16 + lo] = O1[v] * inv;
  }
}

// ---------------- Kernel 3: output projections + gating + LayerNorm ------------
__global__ __launch_bounds__(256) void out_gate_ln(
    const float* __restrict__ ctx, const float* __restrict__ M,
    const float* __restrict__ Wo, const float* __restrict__ bo,
    const float* __restrict__ Wg, const float* __restrict__ bg,
    const float* __restrict__ Wb, const float* __restrict__ bb,
    const float* __restrict__ gamma, const float* __restrict__ beta,
    const float* __restrict__ sens, float* __restrict__ out)
{
  __shared__ float cs[16 * 132];   // 16 ctx rows, pad 132 -> conflict-free
  const int tid = threadIdx.x;
#pragma unroll
  for (int j = 0; j < 8; ++j) {
    const int lin = tid + j * 256;
    cs[(lin >> 7) * 132 + (lin & 127)] = ctx[(size_t)blockIdx.x * 16 * DM + lin];
  }
  __syncthreads();
  const int tl    = tid >> 4;    // local token (16 per block)
  const int oi    = tid & 15;    // output element
  const int token = blockIdx.x * 16 + tl;
  float dv = bo[oi], gv = bg[oi], bv = bb[oi];
  const float* cr = cs + tl * 132;
#pragma unroll 4
  for (int i = 0; i < DM; ++i) {
    const float x = cr[i];
    dv = fmaf(x, Wo[i * 16 + oi], dv);
    gv = fmaf(x, Wg[i * 16 + oi], gv);
    bv = fmaf(x, Wb[i * 16 + oi], bv);
  }
  const float g  = sens[oi] / (1.f + __expf(-gv));
  const float m0 = M[(size_t)token * 16 + oi];
  const float mu = m0 * (1.f - g) + dv * g + 0.1f * tanhf(bv);
  const float mean = sum16(mu) * (1.f / 16.f);
  const float var  = sum16((mu - mean) * (mu - mean)) * (1.f / 16.f);
  out[(size_t)token * 16 + oi] =
      (mu - mean) * rsqrtf(var + 1e-6f) * gamma[oi] + beta[oi];
}

// ---------------- launch --------------------------------------------------------
extern "C" void kernel_launch(void* const* d_in, const int* in_sizes, int n_in,
                              void* d_out, int out_size, void* d_ws, size_t ws_size,
                              hipStream_t stream) {
  const float* M  = (const float*)d_in[0];
  const float* Wq = (const float*)d_in[1];
  const float* bq = (const float*)d_in[2];
  const float* Wk = (const float*)d_in[3];
  const float* bk = (const float*)d_in[4];
  const float* Wv = (const float*)d_in[5];
  const float* bv = (const float*)d_in[6];
  const float* Wo = (const float*)d_in[7];
  const float* bo = (const float*)d_in[8];
  const float* Wg = (const float*)d_in[9];
  const float* bg = (const float*)d_in[10];
  const float* Wb = (const float*)d_in[11];
  const float* bb = (const float*)d_in[12];
  const float* gm = (const float*)d_in[13];
  const float* bt = (const float*)d_in[14];
  const float* sm = (const float*)d_in[15];
  float* out = (float*)d_out;

  char* ws = (char*)d_ws;
  const size_t qsz = (size_t)NTOK * DM * sizeof(_Float16);   // 2 MiB per tensor
  _Float16* Qb  = (_Float16*)(ws);
  _Float16* Kb  = (_Float16*)(ws + qsz);
  _Float16* Vb  = (_Float16*)(ws + 2 * qsz);
  float*    ctx = (float*)(ws + 3 * qsz);                    // 4 MiB

  qkv_proj<<<NTOK, 128, 0, stream>>>(M, Wq, bq, Wk, bk, Wv, bv, Qb, Kb, Vb);
  flash_attn<<<512, 128, 0, stream>>>(Qb, Kb, Vb, ctx);
  out_gate_ln<<<NTOK / 16, 256, 0, stream>>>(ctx, M, Wo, bo, Wg, bg, Wb, bb,
                                             gm, bt, sm, out);
}
